// RNNLikeISTA_23390391894562
// MI455X (gfx1250) — matrix operand order repaired and checked by reading, not compiled, not run on hardware
//
#include <hip/hip_runtime.h>

typedef float    v2f  __attribute__((ext_vector_type(2)));
typedef float    v8f  __attribute__((ext_vector_type(8)));
typedef _Float16 v4h  __attribute__((ext_vector_type(4)));
typedef _Float16 v8h  __attribute__((ext_vector_type(8)));
typedef _Float16 v16h __attribute__((ext_vector_type(16)));

#define LAMBDA  0.1f
#define NITER   20
#define LSIG    4096
#define BROWS   4096
#define KTAPS   64
#define HID     256
#define RPB     16                  // rows per block
#define XS      4164                // LDS row stride (floats): 32 pad + 4096 + 36 pad
#define CHUNK   128                 // cols per sub-tile pass (8 waves x 16)
#define MEGA    2048                // cols per write-back granule
#define NSUB    (MEGA / CHUNK)      // 16
#define NMEGA   (LSIG / MEGA)       // 2
#define SPRIME  80                  // padded Toeplitz extent
#define KSTEP   (SPRIME / 4)        // 20 WMMA k-steps of K=4
#define BOFF    (RPB * XS)          // Bmat offset (floats)
#define HOFF    (BOFF + SPRIME*16)  // halo offset (floats)
#define XH      4104                // f16 LDS row stride for MLP

// ---------------------------------------------------------------------------
// ISTA: 16 rows resident in LDS. conv = Toeplitz GEMM via V_WMMA_F32_16X16X4_F32.
// Inner loop: 1 ds_load (b64 pair, immediate offsets) + 1 wmma per K-step.
// y staged in / x drained out with async global<->LDS DMA (ASYNCcnt path).
// ---------------------------------------------------------------------------
__global__ __launch_bounds__(256, 1)
void ista_kernel(const float* __restrict__ y,
                 const float* __restrict__ pulse,
                 float* __restrict__ xout,
                 float* __restrict__ Bt) {
    extern __shared__ float S[];
    const int tid = threadIdx.x;
    const int m0  = blockIdx.x * RPB;

    // stage y rows via async global->LDS (no VGPR round-trip); S base is LDS 0
    for (int i = tid; i < RPB * (LSIG / 4); i += 256) {
        int r = i >> 10, c4 = i & 1023;
        const float* gp = y + (size_t)(m0 + r) * LSIG + 4 * c4;
        int ldsOff = (r * XS + 32 + 4 * c4) * 4;
        asm volatile("global_load_async_to_lds_b128 %0, %1, off"
                     :: "v"(ldsOff), "v"(gp) : "memory");
    }

    float Lc = 0.f;
    for (int t = 0; t < KTAPS; ++t) { float p = pulse[t]; Lc += p * p; }
    const float invL  = 1.0f / Lc;
    const float theta = LAMBDA / Lc;

    // Toeplitz B: Bm[s][n] = k[s-n-1], k[t] = pulse[63-t]
    for (int i = tid; i < SPRIME * 16; i += 256) {
        int s = i >> 4, n = i & 15, t = s - n - 1;
        S[BOFF + i] = (t >= 0 && t < KTAPS) ? pulse[KTAPS - 1 - t] : 0.f;
    }
    for (int i = tid; i < RPB * (32 + 36); i += 256) {
        int r = i / 68, p = i % 68;
        int col = (p < 32) ? p : (32 + LSIG + (p - 32));
        S[r * XS + col] = 0.f;
    }
    asm volatile("s_wait_asynccnt 0" ::: "memory");
    __syncthreads();

    const int lane = tid & 31, w = tid >> 5, hi = lane >> 4, r16 = lane & 15;

    // hoist B operand into registers: lane needs K = 4*s0 + 2*hi + {0,1}
    float breg[2 * KSTEP];
#pragma unroll
    for (int s0 = 0; s0 < KSTEP; ++s0) {
        breg[2 * s0 + 0] = S[BOFF + (4 * s0 + 2 * hi + 0) * 16 + r16];
        breg[2 * s0 + 1] = S[BOFF + (4 * s0 + 2 * hi + 1) * 16 + r16];
    }

#pragma unroll 1
    for (int it = 0; it < NITER; ++it) {
#pragma unroll 1
        for (int mc = 0; mc < NMEGA; ++mc) {
            const int  jm      = mc * MEGA;
            const bool haloAct = (mc != 0);
            float xn[NSUB * 8];

#pragma unroll
            for (int sub = 0; sub < NSUB; ++sub) {
                const int c0 = jm + sub * CHUNK + w * 16;
                v8f acc0 = {0.f,0.f,0.f,0.f,0.f,0.f,0.f,0.f};
                v8f acc1 = {0.f,0.f,0.f,0.f,0.f,0.f,0.f,0.f};
#pragma unroll
                for (int s0 = 0; s0 < KSTEP; ++s0) {
                    const int colBase = c0 - 32 + 4 * s0 + 2 * hi;   // even -> aligned pair
                    int idx = r16 * XS + 32 + colBase;
                    if (sub == 0 && s0 < 8) {                        // only seam can hit halo
                        int idxH = HOFF + r16 * 32 + (colBase - jm + 32);
                        idx = (haloAct && colBase < jm) ? idxH : idx;
                    }
                    v2f a = *(const v2f*)&S[idx];
                    v2f b; b.x = breg[2 * s0]; b.y = breg[2 * s0 + 1];
                    if (s0 & 1)
                        acc1 = __builtin_amdgcn_wmma_f32_16x16x4_f32(
                                   false, a, false, b, (short)0, acc1, false, false);
                    else
                        acc0 = __builtin_amdgcn_wmma_f32_16x16x4_f32(
                                   false, a, false, b, (short)0, acc0, false, false);
                }

                union { v8f v; float f[8]; } A0, A1; A0.v = acc0; A1.v = acc1;
                const int col = c0 + r16;
                // tile-major B_term: producer thread == consumer thread
                const size_t gbase =
                    (((size_t)blockIdx.x * (NMEGA * NSUB * 8))
                     + (size_t)((mc * NSUB + sub) * 8 + w)) * 256 + (size_t)lane * 8;
                if (it == 0) {
                    float bt[8];
#pragma unroll
                    for (int vv = 0; vv < 8; ++vv) {
                        bt[vv] = (A0.f[vv] + A1.f[vv]) * invL;
                        xn[sub * 8 + vv] = copysignf(fmaxf(fabsf(bt[vv]) - theta, 0.f), bt[vv]);
                    }
                    *(float4*)&Bt[gbase]     = make_float4(bt[0], bt[1], bt[2], bt[3]);
                    *(float4*)&Bt[gbase + 4] = make_float4(bt[4], bt[5], bt[6], bt[7]);
                } else {
                    float4 b0 = *(const float4*)&Bt[gbase];
                    float4 b1 = *(const float4*)&Bt[gbase + 4];
                    float btv[8] = {b0.x, b0.y, b0.z, b0.w, b1.x, b1.y, b1.z, b1.w};
#pragma unroll
                    for (int vv = 0; vv < 8; ++vv) {
                        const int m = vv + hi * 8;
                        float c = btv[vv] + S[m * XS + 32 + col]
                                - (A0.f[vv] + A1.f[vv]) * invL;
                        xn[sub * 8 + vv] = copysignf(fmaxf(fabsf(c) - theta, 0.f), c);
                    }
                }
            }
            __syncthreads();                       // all conv reads done
            if (mc == 0) {                         // save old seam cols [MEGA-32, MEGA)
                for (int i = tid; i < RPB * 32; i += 256) {
                    int rr = i >> 5, cc = i & 31;
                    S[HOFF + i] = S[rr * XS + 32 + MEGA - 32 + cc];
                }
            }
            __syncthreads();                       // seam saved before overwrite
#pragma unroll
            for (int sub = 0; sub < NSUB; ++sub) {
                const int col = jm + sub * CHUNK + w * 16 + r16;
#pragma unroll
                for (int vv = 0; vv < 8; ++vv)
                    S[(vv + hi * 8) * XS + 32 + col] = xn[sub * 8 + vv];
            }
            __syncthreads();                       // writes visible
        }
    }

    // drain x via async LDS->global DMA
    for (int i = tid; i < RPB * (LSIG / 4); i += 256) {
        int r = i >> 10, c4 = i & 1023;
        float* gp = xout + (size_t)(m0 + r) * LSIG + 4 * c4;
        int ldsOff = (r * XS + 32 + 4 * c4) * 4;
        asm volatile("global_store_async_from_lds_b128 %0, %1, off"
                     :: "v"(gp), "v"(ldsOff) : "memory");
    }
    asm volatile("s_wait_asynccnt 0" ::: "memory");
}

// ---------------------------------------------------------------------------
// W1 transpose + f16 cast: W1t[n, k] = (f16) W1[k, n]
// ---------------------------------------------------------------------------
__global__ void transpose_w1(const float* __restrict__ W1, _Float16* __restrict__ W1t) {
    int id = blockIdx.x * blockDim.x + threadIdx.x;
    int n = id & (HID - 1);
    int k = id >> 8;
    W1t[(size_t)n * LSIG + k] = (_Float16)W1[id];
}

__global__ void init_peak(const float* __restrict__ b2, float* __restrict__ peak) {
    int i = blockIdx.x * blockDim.x + threadIdx.x;
    if (i < BROWS) peak[i] = b2[0] * (float)LSIG;
}

// ---------------------------------------------------------------------------
// MLP: relu(x@W1 + b1) @ W2 * L, fused; v_wmma_f32_16x16x32_f16.
// ---------------------------------------------------------------------------
__global__ __launch_bounds__(256)
void mlp_kernel(const float* __restrict__ xg,
                const _Float16* __restrict__ W1t,
                const float* __restrict__ b1,
                const float* __restrict__ W2,
                float* __restrict__ peak) {
    extern __shared__ _Float16 Xh[];
    const int tid = threadIdx.x;
    const int m0  = blockIdx.x * RPB;

    for (int i = tid; i < RPB * (LSIG / 4); i += 256) {
        int r = i >> 10, c4 = i & 1023;
        float4 v = ((const float4*)(xg + (size_t)(m0 + r) * LSIG))[c4];
        v4h h; h.x = (_Float16)v.x; h.y = (_Float16)v.y;
               h.z = (_Float16)v.z; h.w = (_Float16)v.w;
        *(v4h*)&Xh[r * XH + 4 * c4] = h;
    }
    __syncthreads();

    const int lane = tid & 31, w = tid >> 5, hi = lane >> 4, r16 = lane & 15;
    const int n0a = w * 32, n0b = n0a + 16;

    v8f accA = {0.f,0.f,0.f,0.f,0.f,0.f,0.f,0.f};
    v8f accB = {0.f,0.f,0.f,0.f,0.f,0.f,0.f,0.f};
    const _Float16* WrA = W1t + (size_t)(n0a + r16) * LSIG;
    const _Float16* WrB = W1t + (size_t)(n0b + r16) * LSIG;

#pragma unroll 4
    for (int k0 = 0; k0 < LSIG; k0 += 32) {
        v16h a, bA, bB;
        ((v8h*)&a)[0]  = *(const v8h*)&Xh[r16 * XH + k0 + hi * 16];
        ((v8h*)&a)[1]  = *(const v8h*)&Xh[r16 * XH + k0 + hi * 16 + 8];
        ((v8h*)&bA)[0] = *(const v8h*)&WrA[k0 + hi * 16];
        ((v8h*)&bA)[1] = *(const v8h*)&WrA[k0 + hi * 16 + 8];
        ((v8h*)&bB)[0] = *(const v8h*)&WrB[k0 + hi * 16];
        ((v8h*)&bB)[1] = *(const v8h*)&WrB[k0 + hi * 16 + 8];
        accA = __builtin_amdgcn_wmma_f32_16x16x32_f16(
                   false, a, false, bA, (short)0, accA, false, false);
        accB = __builtin_amdgcn_wmma_f32_16x16x32_f16(
                   false, a, false, bB, (short)0, accB, false, false);
    }

#pragma unroll
    for (int t = 0; t < 2; ++t) {
        const int   n  = (t == 0 ? n0a : n0b) + r16;
        const float w2 = W2[n] * (float)LSIG;
        const float bb = b1[n];
        union { v8f v; float f[8]; } U; U.v = (t == 0) ? accA : accB;
#pragma unroll
        for (int vv = 0; vv < 8; ++vv) {
            float f = fmaxf(U.f[vv] + bb, 0.f);
            float p = f * w2;
            p += __shfl_xor(p, 1, 32);
            p += __shfl_xor(p, 2, 32);
            p += __shfl_xor(p, 4, 32);
            p += __shfl_xor(p, 8, 32);
            if (r16 == 0) atomicAdd(&peak[m0 + vv + hi * 8], p);
        }
    }
}

extern "C" void kernel_launch(void* const* d_in, const int* in_sizes, int n_in,
                              void* d_out, int out_size, void* d_ws, size_t ws_size,
                              hipStream_t stream) {
    const float* y     = (const float*)d_in[0];
    const float* pulse = (const float*)d_in[1];
    const float* W1    = (const float*)d_in[2];
    const float* b1    = (const float*)d_in[3];
    const float* W2    = (const float*)d_in[4];
    const float* b2    = (const float*)d_in[5];

    float* out  = (float*)d_out;
    float* xout = out;
    float* peak = out + (size_t)BROWS * LSIG;

    float*    Bt  = (float*)d_ws;
    _Float16* W1t = (_Float16*)((char*)d_ws + (size_t)BROWS * LSIG * 4);

    const size_t istaLds = (size_t)(RPB * XS + SPRIME * 16 + RPB * 32) * sizeof(float);
    const size_t mlpLds  = (size_t)RPB * XH * sizeof(_Float16);
    (void)hipFuncSetAttribute(reinterpret_cast<const void*>(ista_kernel),
                              hipFuncAttributeMaxDynamicSharedMemorySize, (int)istaLds);
    (void)hipFuncSetAttribute(reinterpret_cast<const void*>(mlp_kernel),
                              hipFuncAttributeMaxDynamicSharedMemorySize, (int)mlpLds);

    transpose_w1<<<(LSIG * HID) / 256, 256, 0, stream>>>(W1, W1t);
    ista_kernel<<<BROWS / RPB, 256, istaLds, stream>>>(y, pulse, xout, Bt);
    init_peak<<<(BROWS + 255) / 256, 256, 0, stream>>>(b2, peak);
    mlp_kernel<<<BROWS / RPB, 256, mlpLds, stream>>>(xout, W1t, b1, W2, peak);
}
